// SpatialSumLayer_27084063769180
// MI455X (gfx1250) — matrix operand order repaired and checked by reading, compiled, not run on hardware
//
#include <hip/hip_runtime.h>
#include <hip/hip_bf16.h>

typedef __attribute__((ext_vector_type(16))) _Float16 v16h;
typedef __attribute__((ext_vector_type(8)))  _Float16 v8h;
typedef __attribute__((ext_vector_type(8)))  float    v8f;
typedef __attribute__((ext_vector_type(4)))  float    v4f;

#define HW   1024   // 32*32 spatial
#define CIN  32
#define COUT 64
#define B    32

#define LOG2E 1.4426950408889634f
#define LN2   0.6931471805599453f

// Raw v_exp_f32 / v_log_f32 (no denorm fixups; safe — see range analysis):
// inputs to exp are <= 0 (post max-subtraction), results >= ~1e-5 or flushed
// identically by the f16 convert; log arguments are >= ~1e-4 (softmax term at
// the argmax channel has E==1), far above f32 denormals.
__device__ __forceinline__ float fast_exp(float x) {
  return __builtin_amdgcn_exp2f(x * LOG2E);
}
__device__ __forceinline__ float fast_log(float x) {
  return __builtin_amdgcn_logf(x) * LN2;
}

// ---------------------------------------------------------------------------
// Kernel 1: E[hw][b][ci] = exp(x[b][ci][hw] - m[b][hw])  (f16), m -> mbuf[hw][b]
// One thread per (b, hw): 32768 threads. Reads coalesced over hw.
// ---------------------------------------------------------------------------
__global__ __launch_bounds__(256) void exp_x_kernel(
    const float* __restrict__ x, _Float16* __restrict__ Ebuf,
    float* __restrict__ mbuf) {
  int t  = blockIdx.x * blockDim.x + threadIdx.x;   // [0, 32768)
  int hw = t & (HW - 1);
  int b  = t >> 10;

  const float* xp = x + (size_t)b * CIN * HW + hw;  // stride HW between ci
  float vals[CIN];
  float m = -3.402823466e38f;
#pragma unroll
  for (int ci = 0; ci < CIN; ++ci) {
    vals[ci] = xp[(size_t)ci * HW];
    m = fmaxf(m, vals[ci]);
  }
  mbuf[hw * B + b] = m;

  _Float16* ep = Ebuf + (size_t)hw * (B * CIN) + b * CIN;  // 64B-aligned row
  v8h chunk[4];
#pragma unroll
  for (int ci = 0; ci < CIN; ++ci)
    chunk[ci >> 3][ci & 7] = (_Float16)fast_exp(vals[ci] - m);
#pragma unroll
  for (int k = 0; k < 4; ++k)
    *(v8h*)(ep + k * 8) = chunk[k];                  // global_store_b128
}

// ---------------------------------------------------------------------------
// Kernel 2: P[hw][co][ci] = softmax_ci(weight[co][ci][hw])  (f16)
// One thread per (co, hw): 65536 threads.
// ---------------------------------------------------------------------------
__global__ __launch_bounds__(256) void softmax_w_kernel(
    const float* __restrict__ w, _Float16* __restrict__ Pbuf) {
  int t  = blockIdx.x * blockDim.x + threadIdx.x;   // [0, 65536)
  int hw = t & (HW - 1);
  int co = t >> 10;

  const float* wp = w + (size_t)co * CIN * HW + hw;
  float vals[CIN];
  float m = -3.402823466e38f;
#pragma unroll
  for (int ci = 0; ci < CIN; ++ci) {
    vals[ci] = wp[(size_t)ci * HW];
    m = fmaxf(m, vals[ci]);
  }
  float s = 0.f;
#pragma unroll
  for (int ci = 0; ci < CIN; ++ci) {
    vals[ci] = fast_exp(vals[ci] - m);
    s += vals[ci];
  }
  float inv = __frcp_rn(s);

  _Float16* pp = Pbuf + (size_t)hw * (COUT * CIN) + co * CIN;
  v8h chunk[4];
#pragma unroll
  for (int ci = 0; ci < CIN; ++ci)
    chunk[ci >> 3][ci & 7] = (_Float16)(vals[ci] * inv);
#pragma unroll
  for (int k = 0; k < 4; ++k)
    *(v8h*)(pp + k * 8) = chunk[k];
}

// ---------------------------------------------------------------------------
// Kernel 3: one wave per spatial site. D[b][co] = E[b][:] @ P[:][co] via
// v_wmma_f32_16x16x32_f16 (K = Cin = 32 -> single WMMA per 16x16 tile),
// then out = log(acc) + m.
// C/D layout follows CDNA5 ISA §7.12.2 (wave32); A/B use this code's contiguous k order.
// ---------------------------------------------------------------------------
__global__ __launch_bounds__(256) void lse_wmma_kernel(
    const _Float16* __restrict__ Ebuf, const _Float16* __restrict__ Pbuf,
    const float* __restrict__ mbuf, float* __restrict__ out) {
  int wave = threadIdx.x >> 5;
  int lane = threadIdx.x & 31;
  int hw   = blockIdx.x * 8 + wave;       // 128 blocks * 8 waves = 1024 sites
  int hi   = lane >> 4;                   // half-wave select
  int lr   = lane & 15;                   // row (A: M, B/C/D: N) within tile

  // --- A fragments: 16x32 tile in this code's contiguous k order, shared with B.
  // lane group hi: element e -> K = hi*16 + e
  // => two contiguous 8-f16 chunks at ci = hi*16 and ci = hi*16 + 8.
  v16h afr[2];
#pragma unroll
  for (int mt = 0; mt < 2; ++mt) {
    const _Float16* base =
        Ebuf + (size_t)hw * (B * CIN) + (mt * 16 + lr) * CIN;
    v8h lo = *(const v8h*)(base + hi * 16);        // K = hi*16 .. hi*16+7
    v8h hi8 = *(const v8h*)(base + hi * 16 + 8);  // K = hi*16+8 ..
    v16h f;
#pragma unroll
    for (int e = 0; e < 8; ++e) { f[e] = lo[e]; f[e + 8] = hi8[e]; }
    afr[mt] = f;
  }

  // --- B fragments: 32x16 tile, same contiguous k order as A: lane N = lr, element e -> K = hi*16 + e
  // => 16 contiguous f16 in Pbuf[hw][co][ci] starting at ci = hi*16.
  v16h bfr[4];
#pragma unroll
  for (int nt = 0; nt < 4; ++nt) {
    const _Float16* base =
        Pbuf + (size_t)hw * (COUT * CIN) + (nt * 16 + lr) * CIN + hi * 16;
    v8h lo = *(const v8h*)(base);
    v8h h8 = *(const v8h*)(base + 8);
    v16h f;
#pragma unroll
    for (int e = 0; e < 8; ++e) { f[e] = lo[e]; f[e + 8] = h8[e]; }
    bfr[nt] = f;
  }

  // --- Single per-lane output base; all 64 store addresses are this base
  // plus a compile-time constant (max ((16+7)*64+48)*4096 B = 6.2 MB, which
  // fits the signed 24-bit IOFFSET of global_store_b32 -> zero per-store VALU).
  // C/D layout: lane -> N = lr, VGPR v -> M = v + 8*hi (within 16x16 tile).
  float* obase = out + ((size_t)(hi * 8) * COUT + lr) * HW + hw;

  // --- 8 WMMAs + log epilogue.
#pragma unroll
  for (int mt = 0; mt < 2; ++mt) {
    const float* mb = mbuf + hw * B + mt * 16 + hi * 8;
    v4f m0 = *(const v4f*)(mb);
    v4f m1 = *(const v4f*)(mb + 4);
#pragma unroll
    for (int nt = 0; nt < 4; ++nt) {
      v8f c = {};
      c = __builtin_amdgcn_wmma_f32_16x16x32_f16(
          /*neg_a=*/false, afr[mt], /*neg_b=*/false, bfr[nt],
          /*c_mod=*/(short)0, c, /*reuse_a=*/false, /*reuse_b=*/false);
#pragma unroll
      for (int v = 0; v < 8; ++v) {
        float mv = (v < 4) ? m0[v & 3] : m1[v & 3];
        // constant index: ((mt*16+v)*COUT + nt*16) * HW
        obase[(size_t)((mt * 16 + v) * COUT + nt * 16) * HW] =
            fast_log(c[v]) + mv;
      }
    }
  }
}

extern "C" void kernel_launch(void* const* d_in, const int* in_sizes, int n_in,
                              void* d_out, int out_size, void* d_ws, size_t ws_size,
                              hipStream_t stream) {
  const float* x = (const float*)d_in[0];   // [32, 32, 32, 32]
  const float* w = (const float*)d_in[1];   // [64, 32, 32, 32]
  float* out = (float*)d_out;               // [32, 64, 32, 32]

  char* ws = (char*)d_ws;
  _Float16* Ebuf = (_Float16*)ws;                          // 1024*32*32*2  = 2 MB
  _Float16* Pbuf = (_Float16*)(ws + (2u << 20));           // 1024*64*32*2  = 4 MB
  float*    mbuf = (float*)(ws + (6u << 20));              // 1024*32*4     = 128 KB

  exp_x_kernel    <<<128, 256, 0, stream>>>(x, Ebuf, mbuf);   // 32768 threads
  softmax_w_kernel<<<256, 256, 0, stream>>>(w, Pbuf);         // 65536 threads
  lse_wmma_kernel <<<128, 256, 0, stream>>>(Ebuf, Pbuf, mbuf, out); // 1024 waves
}
